// SSMBlock_73031623901721
// MI455X (gfx1250) — compile-verified, matrix operand order checked by reading, not run on hardware
//
#include <hip/hip_runtime.h>
#include <hip/hip_bf16.h>

typedef __attribute__((ext_vector_type(16))) _Float16 v16h;
typedef __attribute__((ext_vector_type(8)))  _Float16 v8h;
typedef __attribute__((ext_vector_type(8)))  float    v8f;
typedef int i32x4 __attribute__((vector_size(16)));  // matches async-lds builtin pointee

static constexpr int Bb = 4, Tt = 4096, Cc = 512, C4 = 2048, BTt = Bb * Tt;
static constexpr int NS_ITERS = 16;

#ifndef __has_builtin
#define __has_builtin(x) 0
#endif

#if __has_builtin(__builtin_amdgcn_global_load_async_to_lds_b128)
#define HAVE_ASYNC_LDS 1
#else
#define HAVE_ASYNC_LDS 0
#endif

// 16-byte global -> LDS copy. Prefers the gfx1250 async-to-LDS path (ASYNCcnt,
// no VGPR round trip); falls back to load+ds_store if the builtin is absent.
// Builtin signature (from clang diagnostic): (i32x4 AS1*, i32x4 AS3*, imm, imm).
__device__ __forceinline__ void cp16_g2l(const _Float16* g, _Float16* l) {
#if HAVE_ASYNC_LDS
  __builtin_amdgcn_global_load_async_to_lds_b128(
      (__attribute__((address_space(1))) i32x4*)g,
      (__attribute__((address_space(3))) i32x4*)l, 0, 0);
#else
  *(v8h*)l = *(const v8h*)g;
#endif
}

template <int N>
__device__ __forceinline__ void wait_async() {
#if HAVE_ASYNC_LDS
#if __has_builtin(__builtin_amdgcn_s_wait_asynccnt)
  __builtin_amdgcn_s_wait_asynccnt(N);
#else
  if constexpr (N == 0)
    asm volatile("s_wait_asynccnt 0x0" ::: "memory");
  else if constexpr (N == 2)
    asm volatile("s_wait_asynccnt 0x2" ::: "memory");
  else
    asm volatile("s_wait_asynccnt 0x4" ::: "memory");
#endif
#endif
}

// ---------------------------------------------------------------------------
// WMMA GEMM:  D = [Cin +] A * op(B)
//   A  : (M,K) f16 row-major, lda = K
//   NT : Bm = (N,K) f16 row-major (weight "W"), ldb=K  -> D[m,n]=sum_k A[m,k]W[n,k]
//   NN : Bm = (K,N) f16 row-major,              ldb=N  -> D[m,n]=sum_k A[m,k]B[k,n]
// Block: 256 threads = 8 waves; tile 128(M) x 128(N); k-step 32; LDS double-
// buffered; A/B tiles staged with async-to-LDS. Wave (wm 0..3, wn 0..1) owns a
// 32x64 strip = 2x4 WMMA 16x16 subtiles (8 v_wmma per k-step).
// ---------------------------------------------------------------------------
template <bool NT, bool STORE_F16, bool HAS_CIN>
__global__ void __launch_bounds__(256)
k_gemm_wmma(const _Float16* __restrict__ A, const _Float16* __restrict__ Bm,
            const float* __restrict__ Cin, void* __restrict__ Dout,
            int M, int N, int K, int lda, int ldb) {
  constexpr int BM = 128, BN = 128, BK = 32, LDSW = 40;  // 40-half row stride
  __shared__ __align__(16) _Float16 lA[2][BM * LDSW];
  __shared__ __align__(16) _Float16 lB[2][BN * LDSW];

  const int tid  = threadIdx.x;
  const int m0   = blockIdx.x * BM;
  const int n0   = blockIdx.y * BN;
  const int wave = tid >> 5, lane = tid & 31;
  const int wm = wave & 3, wn = wave >> 2;
  const int lrow = lane & 15;
  const int kb   = (lane < 16) ? 0 : 8;  // K sub-offset per 16-bit frag layout

  // C/D layout: VGPR j -> M = j (+8 for lanes 16..31), N = lane%16
  const int mb0 = m0 + wm * 32 + ((lane < 16) ? 0 : 8);
  const int nb0 = n0 + wn * 64 + lrow;

  v8f acc[2][4];
  if constexpr (HAS_CIN) {
#pragma unroll
    for (int j2 = 0; j2 < 2; ++j2)
#pragma unroll
      for (int s = 0; s < 4; ++s)
#pragma unroll
        for (int j = 0; j < 8; ++j)
          acc[j2][s][j] = Cin[(size_t)(mb0 + j2 * 16 + j) * N + (nb0 + s * 16)];
  } else {
#pragma unroll
    for (int j2 = 0; j2 < 2; ++j2)
#pragma unroll
      for (int s = 0; s < 4; ++s)
#pragma unroll
        for (int j = 0; j < 8; ++j) acc[j2][s][j] = 0.0f;
  }

  // Per-thread async issues per stage: NT -> 4 (A:2 + B:2), NN -> 2 (A only).
  constexpr int TAIL = NT ? 4 : 2;

  auto stage = [&](int k0, int buf) {
    {  // A tile 128x32 halves: thread -> row = tid/2, two 16B chunks
      const int row = tid >> 1, cb = (tid & 1) * 2;
      const _Float16* s0 = A + (size_t)(m0 + row) * lda + k0 + cb * 8;
      cp16_g2l(s0, &lA[buf][row * LDSW + cb * 8]);
      cp16_g2l(s0 + 8, &lA[buf][row * LDSW + cb * 8 + 8]);
    }
    if constexpr (NT) {
      const int row = tid >> 1, cb = (tid & 1) * 2;
      const _Float16* s0 = Bm + (size_t)(n0 + row) * ldb + k0 + cb * 8;
      cp16_g2l(s0, &lB[buf][row * LDSW + cb * 8]);
      cp16_g2l(s0 + 8, &lB[buf][row * LDSW + cb * 8 + 8]);
    } else {  // transpose (K,N) -> fragment-major LDS via VGPRs
      const int kk = tid >> 3;
      const int nb = (tid & 7) * 16;
      const v8h b0 = *(const v8h*)(Bm + (size_t)(k0 + kk) * ldb + n0 + nb);
      const v8h b1 = *(const v8h*)(Bm + (size_t)(k0 + kk) * ldb + n0 + nb + 8);
#pragma unroll
      for (int i = 0; i < 8; ++i) {
        lB[buf][(nb + i) * LDSW + kk]     = b0[i];
        lB[buf][(nb + 8 + i) * LDSW + kk] = b1[i];
      }
    }
  };

  const int nk = K / BK;
  stage(0, 0);
  for (int i = 0; i < nk; ++i) {
    const int buf = i & 1;
    if (i + 1 < nk) {
      stage((i + 1) * BK, buf ^ 1);  // prefill next buffer before waiting
      wait_async<TAIL>();            // async loads retire in order per thread
    } else {
      wait_async<0>();
    }
    __syncthreads();

    v16h af[2];
#pragma unroll
    for (int j2 = 0; j2 < 2; ++j2) {
      const int r = (wm * 32 + j2 * 16 + lrow) * LDSW;
      const v8h a0 = *(const v8h*)&lA[buf][r + kb];
      const v8h a1 = *(const v8h*)&lA[buf][r + kb + 16];
      af[j2] = __builtin_shufflevector(a0, a1, 0, 1, 2, 3, 4, 5, 6, 7, 8, 9, 10,
                                       11, 12, 13, 14, 15);
    }
#pragma unroll
    for (int s = 0; s < 4; ++s) {
      const int c = (wn * 64 + s * 16 + lrow) * LDSW;
      const v8h b0 = *(const v8h*)&lB[buf][c + kb];
      const v8h b1 = *(const v8h*)&lB[buf][c + kb + 16];
      const v16h bf = __builtin_shufflevector(b0, b1, 0, 1, 2, 3, 4, 5, 6, 7, 8,
                                              9, 10, 11, 12, 13, 14, 15);
#pragma unroll
      for (int j2 = 0; j2 < 2; ++j2)
        acc[j2][s] = __builtin_amdgcn_wmma_f32_16x16x32_f16(
            false, af[j2], false, bf, (short)0, acc[j2][s], false, false);
    }
    __syncthreads();
  }

#pragma unroll
  for (int j2 = 0; j2 < 2; ++j2)
#pragma unroll
    for (int s = 0; s < 4; ++s)
#pragma unroll
      for (int j = 0; j < 8; ++j) {
        const size_t off = (size_t)(mb0 + j2 * 16 + j) * N + (nb0 + s * 16);
        if constexpr (STORE_F16)
          ((_Float16*)Dout)[off] = (_Float16)acc[j2][s][j];
        else
          ((float*)Dout)[off] = acc[j2][s][j];
      }
}

// ---------------------------------------------------------------------------
// LayerNorm over C=512, f16 output. One block (256 thr) per token.
// ---------------------------------------------------------------------------
__global__ void __launch_bounds__(256)
k_layernorm_f16(const float* __restrict__ x, const float* __restrict__ w,
                _Float16* __restrict__ h) {
  __shared__ float s1[256], s2[256];
  const size_t base = (size_t)blockIdx.x * Cc;
  const int tid = threadIdx.x;
  const float e0 = x[base + tid], e1 = x[base + tid + 256];
  s1[tid] = e0 + e1;
  s2[tid] = e0 * e0 + e1 * e1;
  __syncthreads();
  for (int st = 128; st > 0; st >>= 1) {
    if (tid < st) { s1[tid] += s1[tid + st]; s2[tid] += s2[tid + st]; }
    __syncthreads();
  }
  const float mu  = s1[0] * (1.0f / Cc);
  const float var = s2[0] * (1.0f / Cc) - mu * mu;
  const float rs  = rsqrtf(var + 1e-5f);
  h[base + tid]       = (_Float16)((e0 - mu) * rs * w[tid]);
  h[base + tid + 256] = (_Float16)((e1 - mu) * rs * w[tid + 256]);
}

// RMS-normalize acc row and blend into q (only tokens with t>=d change).
__global__ void __launch_bounds__(256)
k_rms_blend(const float* __restrict__ acc, _Float16* __restrict__ q, int d) {
  const int tok = blockIdx.x;
  const int t = tok & (Tt - 1);
  if (t < d) return;
  __shared__ float s2[256];
  const size_t base = (size_t)tok * Cc;
  const int tid = threadIdx.x;
  const float a0 = acc[base + tid], a1 = acc[base + tid + 256];
  s2[tid] = a0 * a0 + a1 * a1;
  __syncthreads();
  for (int st = 128; st > 0; st >>= 1) {
    if (tid < st) s2[tid] += s2[tid + st];
    __syncthreads();
  }
  const float rs = rsqrtf(s2[0] * (1.0f / Cc) + 1e-6f);
  q[base + tid]       = (_Float16)(a0 * rs);
  q[base + tid + 256] = (_Float16)(a1 * rs);
}

// left[b,t,:] = (t>=d) ? q[b,t-d,:] : identity
__global__ void __launch_bounds__(256)
k_build_left(const _Float16* __restrict__ q, const float* __restrict__ iden,
             _Float16* __restrict__ left, int d, long long n) {
  const long long idx = (long long)blockIdx.x * 256 + threadIdx.x;
  if (idx >= n) return;
  const int c = (int)(idx & (Cc - 1));
  const long long bt = idx >> 9;
  const int t = (int)(bt & (Tt - 1));
  left[idx] = (t >= d) ? q[idx - (long long)d * Cc] : (_Float16)iden[c];
}

__global__ void __launch_bounds__(256)
k_convert_f16(const float* __restrict__ in, _Float16* __restrict__ out,
              long long n) {
  const long long i = (long long)blockIdx.x * 256 + threadIdx.x;
  if (i < n) out[i] = (_Float16)in[i];
}

__global__ void __launch_bounds__(256)
k_mul_qv(const _Float16* __restrict__ q, const _Float16* __restrict__ v,
         _Float16* __restrict__ u, long long n) {
  const long long i = (long long)blockIdx.x * 256 + threadIdx.x;
  if (i < n) u[i] = (_Float16)((float)q[i] * (float)v[i]);
}

__global__ void __launch_bounds__(256)
k_gelu_f16(const float* __restrict__ in, _Float16* __restrict__ out,
           long long n) {
  const long long i = (long long)blockIdx.x * 256 + threadIdx.x;
  if (i < n) {
    const float xv = in[i];
    out[i] = (_Float16)(0.5f * xv * (1.0f + erff(xv * 0.70710678118654752f)));
  }
}

// X0 = W / ||W||_F  (guarantees ||X0||_2 < 1 so Newton-Schulz converges)
__global__ void __launch_bounds__(1024)
k_frob_init(const float* __restrict__ W, _Float16* __restrict__ Xh, int n) {
  __shared__ float red[1024];
  float s = 0.0f;
  for (int i = threadIdx.x; i < n; i += 1024) { const float v = W[i]; s += v * v; }
  red[threadIdx.x] = s;
  __syncthreads();
  for (int st = 512; st > 0; st >>= 1) {
    if (threadIdx.x < st) red[threadIdx.x] += red[threadIdx.x + st];
    __syncthreads();
  }
  const float sc = rsqrtf(red[0] + 1e-12f);
  for (int i = threadIdx.x; i < n; i += 1024) Xh[i] = (_Float16)(W[i] * sc);
}

// T = 1.5 I - 0.5 Y   (Y = X X^T symmetric -> T symmetric, so T@X needs no transposes)
__global__ void __launch_bounds__(256)
k_ns_t(const float* __restrict__ Y, _Float16* __restrict__ Th, long long n) {
  const long long idx = (long long)blockIdx.x * 256 + threadIdx.x;
  if (idx >= n) return;
  const int i = (int)(idx >> 9), j = (int)(idx & (Cc - 1));
  Th[idx] = (_Float16)(((i == j) ? 1.5f : 0.0f) - 0.5f * Y[idx]);
}

// ---------------------------------------------------------------------------
extern "C" void kernel_launch(void* const* d_in, const int* in_sizes, int n_in,
                              void* d_out, int out_size, void* d_ws,
                              size_t ws_size, hipStream_t stream) {
  (void)in_sizes; (void)n_in; (void)out_size; (void)ws_size;
  const float* x    = (const float*)d_in[0];
  const float* ln1w = (const float*)d_in[1];
  const float* Wq   = (const float*)d_in[2];
  const float* Wv   = (const float*)d_in[3];
  const float* Wo   = (const float*)d_in[4];
  const float* iden = (const float*)d_in[5];
  const float* Wp1  = (const float*)d_in[6];
  const float* Wp2  = (const float*)d_in[7];
  const float* ln2w = (const float*)d_in[8];
  const float* Wfc  = (const float*)d_in[9];
  const float* Wpr  = (const float*)d_in[10];
  float* out = (float*)d_out;

  char* cur = (char*)d_ws;
  auto alloc = [&](size_t bytes) -> void* {
    void* p = cur;
    cur += (bytes + 255) & ~(size_t)255;
    return p;
  };
  const long long nAct = (long long)BTt * Cc;   // 8.4M
  const long long nMlp = (long long)BTt * C4;   // 33.5M
  const long long nW   = (long long)Cc * Cc;    // 262144

  _Float16* h16    = (_Float16*)alloc(nAct * 2);
  _Float16* q16    = (_Float16*)alloc(nAct * 2);
  _Float16* v16    = (_Float16*)alloc(nAct * 2);
  _Float16* left16 = (_Float16*)alloc(nAct * 2);  // also reused as u = q*v
  float*    accf   = (float*)alloc(nAct * 4);
  float*    xmid   = (float*)alloc(nAct * 4);
  float*    fcf    = (float*)alloc(nMlp * 4);
  _Float16* fcg16  = (_Float16*)alloc(nMlp * 2);
  _Float16* Wqh  = (_Float16*)alloc(nW * 2);
  _Float16* Wvh  = (_Float16*)alloc(nW * 2);
  _Float16* Woh  = (_Float16*)alloc(nW * 2);
  _Float16* Wfch = (_Float16*)alloc((long long)C4 * Cc * 2);
  _Float16* Wprh = (_Float16*)alloc((long long)Cc * C4 * 2);
  _Float16* Xa1 = (_Float16*)alloc(nW * 2);
  _Float16* Xb1 = (_Float16*)alloc(nW * 2);
  _Float16* Xa2 = (_Float16*)alloc(nW * 2);
  _Float16* Xb2 = (_Float16*)alloc(nW * 2);
  float*    Yf  = (float*)alloc(nW * 4);
  _Float16* Th  = (_Float16*)alloc(nW * 2);

  const dim3 blk(256);
  auto eb = [](long long n) { return dim3((unsigned)((n + 255) / 256)); };

  // 1) h = LN(x, ln1_w) -> f16
  k_layernorm_f16<<<dim3(BTt), blk, 0, stream>>>(x, ln1w, h16);

  // 2) convert weights to f16 once
  k_convert_f16<<<eb(nW), blk, 0, stream>>>(Wq, Wqh, nW);
  k_convert_f16<<<eb(nW), blk, 0, stream>>>(Wv, Wvh, nW);
  k_convert_f16<<<eb(nW), blk, 0, stream>>>(Wo, Woh, nW);
  k_convert_f16<<<eb((long long)C4 * Cc), blk, 0, stream>>>(Wfc, Wfch, (long long)C4 * Cc);
  k_convert_f16<<<eb((long long)Cc * C4), blk, 0, stream>>>(Wpr, Wprh, (long long)Cc * C4);

  // 3) q = h @ Wq^T, v = h @ Wv^T  (f16 outputs)
  const dim3 gAct(BTt / 128, Cc / 128);
  k_gemm_wmma<true, true, false><<<gAct, blk, 0, stream>>>(h16, Wqh, nullptr, q16, BTt, Cc, Cc, Cc, Cc);
  k_gemm_wmma<true, true, false><<<gAct, blk, 0, stream>>>(h16, Wvh, nullptr, v16, BTt, Cc, Cc, Cc, Cc);

  // 4) P1/P2 = polar factors via Newton-Schulz: X <- (1.5 I - 0.5 X X^T) X
  const dim3 gSq(Cc / 128, Cc / 128);
  _Float16* Xs[2][2] = {{Xa1, Xb1}, {Xa2, Xb2}};
  const float* Wp[2] = {Wp1, Wp2};
  _Float16* Ph[2];
  for (int m = 0; m < 2; ++m) {
    _Float16* a = Xs[m][0];
    _Float16* b = Xs[m][1];
    k_frob_init<<<dim3(1), dim3(1024), 0, stream>>>(Wp[m], a, (int)nW);
    for (int it = 0; it < NS_ITERS; ++it) {
      // Y = X X^T   (NT: A=X (M,K), W=X (N,K))
      k_gemm_wmma<true, false, false><<<gSq, blk, 0, stream>>>(a, a, nullptr, Yf, Cc, Cc, Cc, Cc, Cc);
      k_ns_t<<<eb(nW), blk, 0, stream>>>(Yf, Th, nW);
      // X' = T @ X  (NN: A=T (M,K), B=X (K,N) row-major)
      k_gemm_wmma<false, true, false><<<gSq, blk, 0, stream>>>(Th, a, nullptr, b, Cc, Cc, Cc, Cc, Cc);
      _Float16* t = a; a = b; b = t;
    }
    Ph[m] = a;
  }

  // 5) pscan: 12 levels of  q = where(t>=d, rms(left@P1^T + q@P2^T), q)
  for (int d = 1; d < Tt; d <<= 1) {
    k_build_left<<<eb(nAct), blk, 0, stream>>>(q16, iden, left16, d, nAct);
    k_gemm_wmma<true, false, false><<<gAct, blk, 0, stream>>>(left16, Ph[0], nullptr, accf, BTt, Cc, Cc, Cc, Cc);
    k_gemm_wmma<true, false, true><<<gAct, blk, 0, stream>>>(q16, Ph[1], accf, accf, BTt, Cc, Cc, Cc, Cc);
    k_rms_blend<<<dim3(BTt), blk, 0, stream>>>(accf, q16, d);
  }

  // 6) xmid = x + (q*v) @ Wo^T
  k_mul_qv<<<eb(nAct), blk, 0, stream>>>(q16, v16, left16, nAct);
  k_gemm_wmma<true, false, true><<<gAct, blk, 0, stream>>>(left16, Woh, x, xmid, BTt, Cc, Cc, Cc, Cc);

  // 7) MLP: out = xmid + gelu(LN(xmid) @ Wfc^T) @ Wpr^T
  k_layernorm_f16<<<dim3(BTt), blk, 0, stream>>>(xmid, ln2w, h16);
  const dim3 gFc(BTt / 128, C4 / 128);
  k_gemm_wmma<true, false, false><<<gFc, blk, 0, stream>>>(h16, Wfch, nullptr, fcf, BTt, C4, Cc, Cc, Cc);
  k_gelu_f16<<<eb(nMlp), blk, 0, stream>>>(fcf, fcg16, nMlp);
  k_gemm_wmma<true, false, true><<<gAct, blk, 0, stream>>>(fcg16, Wprh, xmid, out, BTt, Cc, C4, C4, C4);
}
